// BiLSTMCRF_15925738733838
// MI455X (gfx1250) — compile-verified, matrix operand order checked by reading, not run on hardware
//
#include <hip/hip_runtime.h>
#include <math.h>

// BiLSTM-CRF loss for MI455X (gfx1250), wave32 + WMMA bf16 + async LDS staging.
//   B=64, S=512, E=1024, HD=512 (gates 4*HD=2048), T=9 tags.
// Pipeline:
//   1) fp32->bf16 conversion of x and LSTM weights (pointwise)
//   2) big WMMA GEMM: Xpre = x @ W_ih^T + (b_ih + b_hh)   (both dirs)
//      - W strip staged in LDS via GLOBAL_LOAD_ASYNC_TO_LDS_B128 (double buffered)
//      - A operand register double-buffered from global
//   3) per timestep (512 iters, both dirs fused per launch):
//        a) WMMA GEMM (register double-buffered): G = Xpre[t] + h_prev @ W_hh^T
//        b) pointwise LSTM cell, h -> bf16 for next step
//   4) emissions = [hf|hb] @ W_tag^T + b_tag  (T=9, VALU kernel)
//   5) CRF numerator + forward algorithm + mean  (1 block, 64 lanes)

#define B_    64
#define S_    512
#define E_    1024
#define HD_   512
#define G4_   2048          // 4*HD
#define T_    9
#define MIN_  (B_*S_)       // 32768 rows of x
#define KCH_  32            // K chunk per WMMA step
#define BSTR_ 40            // LDS row stride (shorts): 32 data + 8 pad, 16B aligned

typedef __attribute__((ext_vector_type(16))) __bf16        v16bf;
typedef __attribute__((ext_vector_type(8)))  float         v8f;
typedef __attribute__((ext_vector_type(4)))  unsigned int  u32x4;
typedef __attribute__((ext_vector_type(4)))  int           v4i;

#if __has_builtin(__builtin_amdgcn_global_load_async_to_lds_b128)
#define HAVE_ASYNC_LDS 1
#else
#define HAVE_ASYNC_LDS 0
#endif

__device__ __forceinline__ void cp_lds_b128(void* lds_dst, const void* g_src) {
#if HAVE_ASYNC_LDS
  // param0: AS1 (global) v4i32*, param1: AS3 (LDS) v4i32*, then imm offset, cpol
  __builtin_amdgcn_global_load_async_to_lds_b128(
      (__attribute__((address_space(1))) v4i*)g_src,
      (__attribute__((address_space(3))) v4i*)lds_dst, 0, 0);
#else
  *(u32x4*)lds_dst = *(const u32x4*)g_src;   // sync fallback (vmem load + ds store)
#endif
}

template<int N>
__device__ __forceinline__ void wait_async() {
#if HAVE_ASYNC_LDS
# if __has_builtin(__builtin_amdgcn_s_wait_asynccnt)
  __builtin_amdgcn_s_wait_asynccnt(N);
# else
  asm volatile("s_wait_asynccnt %0" :: "i"(N) : "memory");
# endif
#endif
}

__device__ __forceinline__ unsigned short f2bf(float f) {
  unsigned int u = __float_as_uint(f);
  unsigned int r = u + 0x7FFFu + ((u >> 16) & 1u);   // round-to-nearest-even
  return (unsigned short)(r >> 16);
}

// ---------------- pointwise helpers ----------------

__global__ void k_f32_to_bf16(const float* __restrict__ src,
                              unsigned short* __restrict__ dst, int n) {
  int i = blockIdx.x * blockDim.x + threadIdx.x;
  if (i < n) dst[i] = f2bf(src[i]);
}

__global__ void k_bias_comb(const float* __restrict__ bi,
                            const float* __restrict__ bh,
                            float* __restrict__ out) {
  int i = blockIdx.x * blockDim.x + threadIdx.x;
  if (i < G4_) out[i] = bi[i] + bh[i];
}

__global__ void k_zero_f32(float* __restrict__ a, int n) {
  int i = blockIdx.x * blockDim.x + threadIdx.x;
  if (i < n) a[i] = 0.0f;
}

__global__ void k_zero_u16(unsigned short* __restrict__ a, int n) {
  int i = blockIdx.x * blockDim.x + threadIdx.x;
  if (i < n) a[i] = 0;
}

// ---------------- WMMA operand loads ----------------
// 16-bit A/B operand layout for v_wmma_*_16x16x32 (ISA 7.12.2):
// lane L: row = L%16, hp = L/16; vector elems 0..7  = [row, k0+8*hp  .. +8)
//                                vector elems 8..15 = [row, k0+16+8*hp .. +8)
// With p = rowbase + hp*8: two contiguous 16B loads at p+k0 and p+k0+16.
__device__ __forceinline__ v16bf ld_op(const unsigned short* p, int k0) {
  union { v16bf v; u32x4 q[2]; } u;
  u.q[0] = *(const u32x4*)(p + k0);
  u.q[1] = *(const u32x4*)(p + k0 + 16);
  return u.v;
}

// Register-double-buffered 16x64 strip GEMM, operands direct from global.
template<int K, int LD>
__device__ __forceinline__ void wave_gemm_16x64_reg(const unsigned short* __restrict__ A, int m0,
                                                    const unsigned short* __restrict__ Bw, int n0,
                                                    v8f* acc, int lr, int hp) {
  const unsigned short* pa = A + (size_t)(m0 + lr) * LD + hp * 8;
  const unsigned short* pb[4];
#pragma unroll
  for (int j = 0; j < 4; ++j)
    pb[j] = Bw + (size_t)(n0 + j * 16 + lr) * LD + hp * 8;

  v16bf a_cur = ld_op(pa, 0);
  v16bf b_cur[4];
#pragma unroll
  for (int j = 0; j < 4; ++j) b_cur[j] = ld_op(pb[j], 0);

  for (int k0 = 0; k0 < K; k0 += KCH_) {
    v16bf a_nxt = a_cur;
    v16bf b_nxt[4] = {b_cur[0], b_cur[1], b_cur[2], b_cur[3]};
    if (k0 + KCH_ < K) {                 // prefetch next iteration into registers
      a_nxt = ld_op(pa, k0 + KCH_);
#pragma unroll
      for (int j = 0; j < 4; ++j) b_nxt[j] = ld_op(pb[j], k0 + KCH_);
    }
#pragma unroll
    for (int j = 0; j < 4; ++j)
      acc[j] = __builtin_amdgcn_wmma_f32_16x16x32_bf16(
          false, a_cur, false, b_cur[j], (short)0, acc[j], false, false);
    a_cur = a_nxt;
#pragma unroll
    for (int j = 0; j < 4; ++j) b_cur[j] = b_nxt[j];
  }
}

// ---------------- big input-projection GEMM ----------------
// C[dir](32768 x 2048) = Xbf(32768 x 1024) @ W[dir]^T + bias[dir]
// grid = (N/64=32, M/128=256, 2 dirs), block = 256 (8 waves stacked along M).
// All 8 waves share the 64 W columns -> stage W chunk (64 x 32 bf16) in LDS via
// async copy, double buffered; A operand register double-buffered from global.
__global__ void __launch_bounds__(256)
k_gemm_in(const unsigned short* __restrict__ X,
          const unsigned short* __restrict__ W0, const unsigned short* __restrict__ W1,
          const float* __restrict__ b0, const float* __restrict__ b1,
          float* __restrict__ C0, float* __restrict__ C1) {
  __shared__ unsigned short sB[2][64 * BSTR_];

  const int tid = threadIdx.x;
  const int lane = tid & 31, wave = tid >> 5;
  const int lr = lane & 15, hp = lane >> 4;
  const int m0 = (blockIdx.y * 8 + wave) * 16;
  const int n0 = blockIdx.x * 64;
  const unsigned short* W = blockIdx.z ? W1 : W0;
  const float* bias       = blockIdx.z ? b1 : b0;
  float* C                = blockIdx.z ? C1 : C0;

  // cooperative copy assignment: 256 threads x 16B = 4KB = one 64x32 bf16 chunk
  const int crow = tid >> 2, cseg = tid & 3;
  const unsigned short* gsrc = W + (size_t)(n0 + crow) * E_ + cseg * 8;
  unsigned short* ldst[2] = { &sB[0][crow * BSTR_ + cseg * 8],
                              &sB[1][crow * BSTR_ + cseg * 8] };

  cp_lds_b128(ldst[0], gsrc);                       // chunk 0 in flight

  const unsigned short* pa = X + (size_t)(m0 + lr) * E_ + hp * 8;
  v16bf a_cur = ld_op(pa, 0);
  v8f acc[4] = {};

  constexpr int NIT = E_ / KCH_;                    // 32 chunks
  for (int i = 0; i < NIT; ++i) {
    const int k0 = i * KCH_;
    v16bf a_nxt = a_cur;
    if (i + 1 < NIT) {
      cp_lds_b128(ldst[(i + 1) & 1], gsrc + k0 + KCH_);   // async fill next buffer
      a_nxt = ld_op(pa, k0 + KCH_);
    }
    if (i + 1 < NIT) wait_async<1>(); else wait_async<0>();  // current chunk landed
    __syncthreads();                                // all waves' copies visible

    const unsigned short* bb = sB[i & 1];
#pragma unroll
    for (int j = 0; j < 4; ++j) {
      v16bf b = ld_op(bb + (j * 16 + lr) * BSTR_ + hp * 8, 0);   // ds_load_b128 x2
      acc[j] = __builtin_amdgcn_wmma_f32_16x16x32_bf16(
          false, a_cur, false, b, (short)0, acc[j], false, false);
    }
    __syncthreads();                                // reads done before buffer reuse
    a_cur = a_nxt;
  }

#pragma unroll
  for (int j = 0; j < 4; ++j) {
    const int n = n0 + j * 16 + lr;
    const float bv = bias[n];
#pragma unroll
    for (int r = 0; r < 8; ++r) {
      const int m = m0 + r + 8 * hp;
      C[(size_t)m * G4_ + n] = acc[j][r] + bv;
    }
  }
}

// ---------------- per-timestep recurrent GEMM ----------------
// G[dir](64 x 2048) = Xpre[dir][rows for step t] + Hprev[dir](64x512) @ Whh[dir]^T
// grid = (4, 4, 2), block = 256  -> 256 waves = 4 Mtiles x 32 Ngroups x 2 dirs
// Small latency-critical GEMM; Whh is L2-resident -> register double-buffer only.
__global__ void __launch_bounds__(256)
k_gemm_step(const unsigned short* __restrict__ H0, const unsigned short* __restrict__ H1,
            const unsigned short* __restrict__ Wh0, const unsigned short* __restrict__ Wh1,
            const float* __restrict__ Xp0, const float* __restrict__ Xp1,
            float* __restrict__ G0, float* __restrict__ G1, int t) {
  const int lane = threadIdx.x & 31, wave = threadIdx.x >> 5;
  const int lr = lane & 15, hp = lane >> 4;
  const int dir = blockIdx.z;
  const int m0 = blockIdx.y * 16;                 // batch tile (0..48)
  const int n0 = (blockIdx.x * 8 + wave) * 64;    // gate-column group
  const unsigned short* H  = dir ? H1  : H0;
  const unsigned short* Wh = dir ? Wh1 : Wh0;
  const float* Xp          = dir ? Xp1 : Xp0;
  float* G                 = dir ? G1  : G0;
  const int s = dir ? (S_ - 1 - t) : t;           // backward dir runs reversed

  v8f acc[4] = {};
  wave_gemm_16x64_reg<HD_, HD_>(H, m0, Wh, n0, acc, lr, hp);

#pragma unroll
  for (int j = 0; j < 4; ++j) {
    const int n = n0 + j * 16 + lr;
#pragma unroll
    for (int r = 0; r < 8; ++r) {
      const int b = m0 + r + 8 * hp;
      const size_t xrow = (size_t)b * S_ + s;     // x is (B,S,E)-ordered
      G[(size_t)b * G4_ + n] = acc[j][r] + Xp[xrow * G4_ + n];
    }
  }
}

// ---------------- pointwise LSTM cell ----------------
__global__ void k_lstm_point(const float* __restrict__ G0, const float* __restrict__ G1,
                             float* __restrict__ Cs0, float* __restrict__ Cs1,
                             unsigned short* __restrict__ Hb0, unsigned short* __restrict__ Hb1,
                             float* __restrict__ Ha0, float* __restrict__ Ha1, int t) {
  int idx = blockIdx.x * blockDim.x + threadIdx.x;
  if (idx >= 2 * B_ * HD_) return;
  const int dir = idx / (B_ * HD_);
  const int rem = idx - dir * (B_ * HD_);
  const int b = rem / HD_, j = rem % HD_;
  const float* G        = dir ? G1  : G0;
  float* Cs             = dir ? Cs1 : Cs0;
  unsigned short* Hbf   = dir ? Hb1 : Hb0;
  float* Hall           = dir ? Ha1 : Ha0;
  const int s = dir ? (S_ - 1 - t) : t;

  const size_t base = (size_t)b * G4_;
  const float gi = G[base + j];
  const float gf = G[base + HD_ + j];
  const float gg = G[base + 2 * HD_ + j];
  const float go = G[base + 3 * HD_ + j];
  const float si = 1.0f / (1.0f + __expf(-gi));
  const float sf = 1.0f / (1.0f + __expf(-gf));
  const float so = 1.0f / (1.0f + __expf(-go));
  const float c  = sf * Cs[rem] + si * tanhf(gg);
  const float h  = so * tanhf(c);
  Cs[rem]  = c;
  Hbf[rem] = f2bf(h);                                   // feeds next step's WMMA
  Hall[((size_t)s * B_ + b) * HD_ + j] = h;             // (S,B,HD) fp32
}

// ---------------- emissions: [hf|hb] @ W_tag^T + b_tag ----------------
__global__ void k_emis(const float* __restrict__ Hf, const float* __restrict__ Hb,
                       const float* __restrict__ Wt, const float* __restrict__ bt,
                       float* __restrict__ em) {
  int idx = blockIdx.x * blockDim.x + threadIdx.x;      // (b*S+s)*T + k
  if (idx >= B_ * S_ * T_) return;
  const int k  = idx % T_;
  const int bs = idx / T_;
  const int s  = bs % S_;
  const int b  = bs / S_;
  const float* hf = Hf + ((size_t)s * B_ + b) * HD_;
  const float* hb = Hb + ((size_t)s * B_ + b) * HD_;
  const float* w  = Wt + (size_t)k * (2 * HD_);
  float acc = bt[k];
  for (int j = 0; j < HD_; ++j) acc += hf[j] * w[j];
  for (int j = 0; j < HD_; ++j) acc += hb[j] * w[HD_ + j];
  em[idx] = acc;                                        // (B,S,T)
}

// ---------------- CRF: gold score + partition + mean ----------------
__global__ void __launch_bounds__(64)
k_crf(const float* __restrict__ em, const int* __restrict__ lab,
      const float* __restrict__ st, const float* __restrict__ et,
      const float* __restrict__ tr, float* __restrict__ out) {
  __shared__ float s_tr[T_ * T_], s_st[T_], s_et[T_], red[B_];
  const int tid = threadIdx.x;
  for (int i = tid; i < T_ * T_; i += 64) s_tr[i] = tr[i];
  if (tid < T_) { s_st[tid] = st[tid]; s_et[tid] = et[tid]; }
  __syncthreads();

  const int b = tid;
  const float* e = em + (size_t)b * S_ * T_;
  const int* l   = lab + (size_t)b * S_;

  // numerator (gold path)
  float num = s_st[l[0]] + e[l[0]];
  for (int s = 1; s < S_; ++s)
    num += s_tr[l[s - 1] * T_ + l[s]] + e[s * T_ + l[s]];
  num += s_et[l[S_ - 1]];

  // forward algorithm (log-alpha)
  float a[T_], na[T_];
  for (int k = 0; k < T_; ++k) a[k] = s_st[k] + e[k];
  for (int s = 1; s < S_; ++s) {
    for (int j = 0; j < T_; ++j) {
      float mx = -1e30f;
      for (int i = 0; i < T_; ++i) mx = fmaxf(mx, a[i] + s_tr[i * T_ + j]);
      float sum = 0.0f;
      for (int i = 0; i < T_; ++i) sum += __expf(a[i] + s_tr[i * T_ + j] - mx);
      na[j] = e[s * T_ + j] + mx + __logf(sum);
    }
    for (int j = 0; j < T_; ++j) a[j] = na[j];
  }
  float mx = -1e30f;
  for (int k = 0; k < T_; ++k) mx = fmaxf(mx, a[k] + s_et[k]);
  float sum = 0.0f;
  for (int k = 0; k < T_; ++k) sum += __expf(a[k] + s_et[k] - mx);
  const float logZ = mx + __logf(sum);

  red[b] = num - logZ;
  __syncthreads();
  if (tid == 0) {
    float tot = 0.0f;
    for (int i = 0; i < B_; ++i) tot += red[i];
    out[0] = -tot / (float)B_;
  }
}

// ---------------- host launch ----------------
extern "C" void kernel_launch(void* const* d_in, const int* in_sizes, int n_in,
                              void* d_out, int out_size, void* d_ws, size_t ws_size,
                              hipStream_t stream) {
  const float* x     = (const float*)d_in[0];
  const int* labels  = (const int*)d_in[1];
  const float* Wih_f = (const float*)d_in[2];
  const float* Whh_f = (const float*)d_in[3];
  const float* bih_f = (const float*)d_in[4];
  const float* bhh_f = (const float*)d_in[5];
  const float* Wih_b = (const float*)d_in[6];
  const float* Whh_b = (const float*)d_in[7];
  const float* bih_b = (const float*)d_in[8];
  const float* bhh_b = (const float*)d_in[9];
  const float* Wtag  = (const float*)d_in[10];
  const float* btag  = (const float*)d_in[11];
  const float* st    = (const float*)d_in[12];
  const float* et    = (const float*)d_in[13];
  const float* tr    = (const float*)d_in[14];
  float* out = (float*)d_out;
  (void)in_sizes; (void)n_in; (void)out_size; (void)ws_size;

  char* ws = (char*)d_ws;
  size_t off = 0;
  auto alloc = [&](size_t bytes) -> char* {
    char* p = ws + off;
    off = (off + bytes + 255) & ~(size_t)255;
    return p;
  };
  unsigned short* xbf  = (unsigned short*)alloc((size_t)MIN_ * E_ * 2);
  unsigned short* wihf = (unsigned short*)alloc((size_t)G4_ * E_ * 2);
  unsigned short* wihb = (unsigned short*)alloc((size_t)G4_ * E_ * 2);
  unsigned short* whhf = (unsigned short*)alloc((size_t)G4_ * HD_ * 2);
  unsigned short* whhb = (unsigned short*)alloc((size_t)G4_ * HD_ * 2);
  float* biasf = (float*)alloc(G4_ * 4);
  float* biasb = (float*)alloc(G4_ * 4);
  float* xpref = (float*)alloc((size_t)MIN_ * G4_ * 4);
  float* xpreb = (float*)alloc((size_t)MIN_ * G4_ * 4);
  float* hallf = (float*)alloc((size_t)S_ * B_ * HD_ * 4);
  float* hallb = (float*)alloc((size_t)S_ * B_ * HD_ * 4);
  unsigned short* hbff = (unsigned short*)alloc(B_ * HD_ * 2);
  unsigned short* hbfb = (unsigned short*)alloc(B_ * HD_ * 2);
  float* csf = (float*)alloc(B_ * HD_ * 4);
  float* csb = (float*)alloc(B_ * HD_ * 4);
  float* g0  = (float*)alloc((size_t)B_ * G4_ * 4);
  float* g1  = (float*)alloc((size_t)B_ * G4_ * 4);
  float* em  = (float*)alloc((size_t)B_ * S_ * T_ * 4);

  // 1) precision conversion
  auto cvt = [&](const float* s_, unsigned short* d_, size_t n) {
    k_f32_to_bf16<<<(unsigned)((n + 255) / 256), 256, 0, stream>>>(s_, d_, (int)n);
  };
  cvt(x,     xbf,  (size_t)MIN_ * E_);
  cvt(Wih_f, wihf, (size_t)G4_ * E_);
  cvt(Wih_b, wihb, (size_t)G4_ * E_);
  cvt(Whh_f, whhf, (size_t)G4_ * HD_);
  cvt(Whh_b, whhb, (size_t)G4_ * HD_);
  k_bias_comb<<<8, 256, 0, stream>>>(bih_f, bhh_f, biasf);
  k_bias_comb<<<8, 256, 0, stream>>>(bih_b, bhh_b, biasb);
  k_zero_f32<<<(B_ * HD_) / 256, 256, 0, stream>>>(csf, B_ * HD_);
  k_zero_f32<<<(B_ * HD_) / 256, 256, 0, stream>>>(csb, B_ * HD_);
  k_zero_u16<<<(B_ * HD_) / 256, 256, 0, stream>>>(hbff, B_ * HD_);
  k_zero_u16<<<(B_ * HD_) / 256, 256, 0, stream>>>(hbfb, B_ * HD_);

  // 2) big input-projection GEMM (both directions)
  k_gemm_in<<<dim3(G4_ / 64, MIN_ / 128, 2), 256, 0, stream>>>(
      xbf, wihf, wihb, biasf, biasb, xpref, xpreb);

  // 3) recurrence: 512 steps, both directions per launch
  for (int t = 0; t < S_; ++t) {
    k_gemm_step<<<dim3(4, 4, 2), 256, 0, stream>>>(
        hbff, hbfb, whhf, whhb, xpref, xpreb, g0, g1, t);
    k_lstm_point<<<(2 * B_ * HD_) / 256, 256, 0, stream>>>(
        g0, g1, csf, csb, hbff, hbfb, hallf, hallb, t);
  }

  // 4) emissions
  const int ne = B_ * S_ * T_;
  k_emis<<<(ne + 255) / 256, 256, 0, stream>>>(hallf, hallb, Wtag, btag, em);

  // 5) CRF loss
  k_crf<<<1, 64, 0, stream>>>(em, labels, st, et, tr, out);
}
